// CausalSelfAttention_52974126629420
// MI455X (gfx1250) — matrix operand order repaired and checked by reading, not compiled, not run on hardware
//
#include <hip/hip_runtime.h>

#define DIM   1024
#define HEADS 16
#define HD    64
#define BATCH 2
#define SEQ   2048
#define ROWS  (BATCH * SEQ)   // 4096

typedef __attribute__((ext_vector_type(16))) __bf16        v16bf;
typedef __attribute__((ext_vector_type(8)))  float         v8f;
typedef __attribute__((ext_vector_type(4)))  unsigned int  v4u;
typedef __attribute__((ext_vector_type(8)))  int           v8i;
typedef __attribute__((ext_vector_type(4)))  int           v4i;

struct alignas(16) I4 { int x, y, z, w; };   // POD 16B chunk (int4 has ctors -> illegal in union)

union Frag {
    v16bf bf;
    I4    q[2];
};

__device__ __forceinline__ unsigned short f2bf(float f) {
    unsigned int u = __float_as_uint(f);
    u += 0x7fffu + ((u >> 16) & 1u);        // round-to-nearest-even
    return (unsigned short)(u >> 16);
}

__device__ __forceinline__ unsigned lds_off(const void* p) {
    // LDS aperture: flat addr low 32 bits == LDS byte offset (ISA 10.2)
    return (unsigned)(unsigned long long)p;
}

// TDM: 2-D bf16 tile load Global -> LDS.  tw = tile width (elems, row-contiguous),
// th = rows, stride = row stride in elems.  D# per ISA 8.3/8.4, higher groups zero (2-D).
__device__ __forceinline__ void tdm_load2d_bf16(unsigned lds, const void* g,
                                                unsigned tw, unsigned th,
                                                unsigned long long stride) {
    unsigned long long ga = (unsigned long long)g;
    v4u g0;
    g0[0] = 1u;                                     // count=1, user descriptor
    g0[1] = lds;                                    // lds_addr
    g0[2] = (unsigned)ga;                           // global_addr[31:0]
    g0[3] = (unsigned)((ga >> 32) & 0x01ffffffull)  // global_addr[56:32]
          | 0x80000000u;                            // type=2 ("image")
    v8i g1;
    g1[0] = 0x10000;                                // data_size=1 (2 bytes/elem)
    g1[1] = (int)((tw & 0xffffu) << 16);            // tensor_dim0[15:0] @bit48
    g1[2] = (int)((tw >> 16) | ((th & 0xffffu) << 16));  // tensor_dim0[31:16] | tensor_dim1[15:0]
    g1[3] = (int)((th >> 16) | (tw << 16));         // tensor_dim1[31:16] | tile_dim0
    g1[4] = (int)(th & 0xffffu);                    // tile_dim1 (tile_dim2=0)
    g1[5] = (int)(unsigned)stride;                  // tensor_dim0_stride[31:0]
    g1[6] = (int)((stride >> 32) & 0xffffull);      // stride[47:32] (dim1_stride=0)
    g1[7] = 0;
    v4i z4; z4[0] = 0; z4[1] = 0; z4[2] = 0; z4[3] = 0;
    v8i z8;
    #pragma unroll
    for (int i = 0; i < 8; ++i) z8[i] = 0;
    __builtin_amdgcn_tensor_load_to_lds(g0, g1, z4, z4, z8, 0);   // 6-arg form (clang-23)
}

// ---------------------------------------------------------------- converts --
__global__ __launch_bounds__(256) void k_cvt(const float* __restrict__ in,
                                             unsigned short* __restrict__ out, int n) {
    int i = blockIdx.x * 256 + threadIdx.x;
    if (i < n) out[i] = f2bf(in[i]);
}

// out[n*nk + k] = bf16(in[k*ncols + n])   (transpose to [N][K] for B-fragments)
__global__ __launch_bounds__(256) void k_cvt_t(const float* __restrict__ in,
                                               unsigned short* __restrict__ out,
                                               int ncols, int nk) {
    int i = blockIdx.x * 256 + threadIdx.x;
    if (i < ncols * nk) {
        int n = i / nk, k = i % nk;
        out[i] = f2bf(in[(size_t)k * ncols + n]);
    }
}

// ------------------------------------------------------------- QKV GEMM -----
// C[4096 x 3072] = xb * wT^T + bias; TDM double-buffered 64x32 A/B tiles.
__global__ __launch_bounds__(128) void k_qkv(const unsigned short* __restrict__ xb,
                                             const unsigned short* __restrict__ wT,
                                             const float* __restrict__ bias,
                                             unsigned short* __restrict__ qb,
                                             unsigned short* __restrict__ kbuf,
                                             unsigned short* __restrict__ vT) {
    __shared__ __align__(16) unsigned short ldsA[2][64 * 32];
    __shared__ __align__(16) unsigned short ldsB[2][64 * 32];
    const int m0 = blockIdx.y * 64, n0 = blockIdx.x * 64;
    const int tid = threadIdx.x;
    const int w = tid >> 5, lane = tid & 31, ml = lane & 15, half = lane >> 4;
    const bool issuer = (__builtin_amdgcn_readfirstlane(tid) >> 5) == 0;  // scalar: wave 0 only

    v8f acc[4];
    #pragma unroll
    for (int t = 0; t < 4; ++t)
        #pragma unroll
        for (int r = 0; r < 8; ++r) acc[t][r] = 0.f;

    if (issuer) {
        tdm_load2d_bf16(lds_off(&ldsA[0][0]), xb + (size_t)m0 * DIM, 32, 64, DIM);
        tdm_load2d_bf16(lds_off(&ldsB[0][0]), wT + (size_t)n0 * DIM, 32, 64, DIM);
    }

    for (int it = 0; it < DIM / 32; ++it) {
        const int cur = it & 1;
        if (issuer) {
            if (it + 1 < DIM / 32) {
                const int kn = (it + 1) * 32;
                tdm_load2d_bf16(lds_off(&ldsA[cur ^ 1][0]), xb + (size_t)m0 * DIM + kn, 32, 64, DIM);
                tdm_load2d_bf16(lds_off(&ldsB[cur ^ 1][0]), wT + (size_t)n0 * DIM + kn, 32, 64, DIM);
                __builtin_amdgcn_s_wait_tensorcnt(2);   // current pair complete (in-order)
            } else {
                __builtin_amdgcn_s_wait_tensorcnt(0);
            }
        }
        __syncthreads();

        Frag a;
        const unsigned short* A = &ldsA[cur][0];
        const unsigned short* B = &ldsB[cur][0];
        const int arow = w * 16 + ml;
        a.q[0] = *(const I4*)&A[arow * 32 + half * 8];
        a.q[1] = *(const I4*)&A[arow * 32 + 16 + half * 8];
        #pragma unroll
        for (int t = 0; t < 4; ++t) {
            Frag b;
            const int brow = t * 16 + ml;
            b.q[0] = *(const I4*)&B[brow * 32 + half * 8];
            b.q[1] = *(const I4*)&B[brow * 32 + 16 + half * 8];
            acc[t] = __builtin_amdgcn_wmma_f32_16x16x32_bf16(
                false, a.bf, false, b.bf, (short)0, acc[t], false, false);
        }
        __syncthreads();
    }

    #pragma unroll
    for (int t = 0; t < 4; ++t) {
        const int Cc = n0 + t * 16 + ml;
        const int three = Cc >> 10, rem = Cc & 1023;
        const int h = rem >> 6, hd = rem & 63;
        const float bv = bias[Cc];
        #pragma unroll
        for (int r = 0; r < 8; ++r) {
            const int R = m0 + w * 16 + r + 8 * half;
            const int b = R >> 11, l = R & 2047;
            const int bh = b * HEADS + h;
            const unsigned short val = f2bf(acc[t][r] + bv);
            if (three == 0)      qb  [((size_t)bh * SEQ + l) * HD + hd] = val;
            else if (three == 1) kbuf[((size_t)bh * SEQ + l) * HD + hd] = val;
            else                 vT  [((size_t)bh * HD + hd) * SEQ + l] = val;  // V transposed
        }
    }
}

// ------------------------------------------------------ flash attention -----
// Per block: 64 query rows (4 waves x 16), TDM double-buffered K(32x64)/V(64x32) tiles.
__global__ __launch_bounds__(128) void k_attn(const unsigned short* __restrict__ qb,
                                              const unsigned short* __restrict__ kbuf,
                                              const unsigned short* __restrict__ vT,
                                              unsigned short* __restrict__ attn) {
    __shared__ __align__(16) unsigned short ldsK[2][32 * 64];
    __shared__ __align__(16) unsigned short ldsV[2][64 * 32];
    __shared__ __align__(16) unsigned short ldsP[4 * 16 * 32];
    const int bh = blockIdx.y;
    const int q0 = blockIdx.x * 64;
    const int tid = threadIdx.x;
    const int w = tid >> 5, lane = tid & 31, ml = lane & 15, half = lane >> 4;
    const int q0w = q0 + w * 16;
    const float scale = 0.125f;   // 1/sqrt(64)
    const bool issuer = (__builtin_amdgcn_readfirstlane(tid) >> 5) == 0;

    // resident Q fragments (16 x 64, two K-steps of 32)
    const unsigned short* qrow = qb + ((size_t)bh * SEQ + q0w + ml) * HD;
    Frag qa[2];
    #pragma unroll
    for (int s = 0; s < 2; ++s) {
        qa[s].q[0] = *(const I4*)(qrow + s * 32 + half * 8);
        qa[s].q[1] = *(const I4*)(qrow + s * 32 + 16 + half * 8);
    }

    float rmax[8], rsum[8];
    v8f o[4];
    #pragma unroll
    for (int r = 0; r < 8; ++r) { rmax[r] = -__builtin_inff(); rsum[r] = 0.f; }
    #pragma unroll
    for (int t = 0; t < 4; ++t)
        #pragma unroll
        for (int r = 0; r < 8; ++r) o[t][r] = 0.f;

    unsigned short* myP = &ldsP[w * 512];
    const int nblocks = q0 / 32 + 2;   // uniform across waves; mask handles upper triangle

    if (issuer) {
        tdm_load2d_bf16(lds_off(&ldsK[0][0]), kbuf + (size_t)bh * SEQ * HD, 64, 32, 64);
        tdm_load2d_bf16(lds_off(&ldsV[0][0]), vT + (size_t)bh * HD * SEQ, 32, 64, SEQ);
    }

    for (int it = 0; it < nblocks; ++it) {
        const int j0 = it * 32;
        const int cur = it & 1;
        if (issuer) {
            if (it + 1 < nblocks) {
                const int jn = j0 + 32;
                tdm_load2d_bf16(lds_off(&ldsK[cur ^ 1][0]),
                                kbuf + ((size_t)bh * SEQ + jn) * HD, 64, 32, 64);
                tdm_load2d_bf16(lds_off(&ldsV[cur ^ 1][0]),
                                vT + (size_t)bh * HD * SEQ + jn, 32, 64, SEQ);
                __builtin_amdgcn_s_wait_tensorcnt(2);
            } else {
                __builtin_amdgcn_s_wait_tensorcnt(0);
            }
        }
        __syncthreads();

        const unsigned short* Kt = &ldsK[cur][0];   // [32 keys][64 dims]
        const unsigned short* Vt = &ldsV[cur][0];   // [64 dims][32 keys]

        // ---- S = Q @ K^T for 32 keys (two 16-wide tiles) ----
        v8f s0, s1;
        #pragma unroll
        for (int r = 0; r < 8; ++r) { s0[r] = 0.f; s1[r] = 0.f; }
        #pragma unroll
        for (int s = 0; s < 2; ++s) {
            Frag b0, b1;
            const unsigned short* kp0 = Kt + ml * 64 + s * 32;
            const unsigned short* kp1 = kp0 + 16 * 64;
            b0.q[0] = *(const I4*)(kp0 + half * 8);
            b0.q[1] = *(const I4*)(kp0 + 16 + half * 8);
            b1.q[0] = *(const I4*)(kp1 + half * 8);
            b1.q[1] = *(const I4*)(kp1 + 16 + half * 8);
            s0 = __builtin_amdgcn_wmma_f32_16x16x32_bf16(
                false, qa[s].bf, false, b0.bf, (short)0, s0, false, false);
            s1 = __builtin_amdgcn_wmma_f32_16x16x32_bf16(
                false, qa[s].bf, false, b1.bf, (short)0, s1, false, false);
        }

        // ---- online softmax (row m = r + 8*half lives in this lane's 16-group) ----
        #pragma unroll
        for (int r = 0; r < 8; ++r) {
            const int Qg = q0w + r + 8 * half;
            float a0 = s0[r] * scale; if (j0 + ml > Qg)      a0 = -__builtin_inff();
            float a1 = s1[r] * scale; if (j0 + 16 + ml > Qg) a1 = -__builtin_inff();
            float mx = fmaxf(a0, a1);
            #pragma unroll
            for (int off = 8; off > 0; off >>= 1)
                mx = fmaxf(mx, __shfl_xor(mx, off, 16));
            const float mnew = fmaxf(rmax[r], mx);
            const float alpha = __expf(rmax[r] - mnew);
            rmax[r] = mnew;
            const float e0 = __expf(a0 - mnew);
            const float e1 = __expf(a1 - mnew);
            float sum = e0 + e1;
            #pragma unroll
            for (int off = 8; off > 0; off >>= 1)
                sum += __shfl_xor(sum, off, 16);
            rsum[r] = rsum[r] * alpha + sum;
            #pragma unroll
            for (int t = 0; t < 4; ++t) o[t][r] *= alpha;
            const int prow = r + 8 * half;
            myP[prow * 32 + ml]      = f2bf(e0);    // re-stripe C-layout -> A-layout
            myP[prow * 32 + 16 + ml] = f2bf(e1);
        }

        // ---- O += P @ V (one K=32 WMMA per 16-wide dim tile) ----
        Frag pa;
        pa.q[0] = *(const I4*)&myP[ml * 32 + half * 8];
        pa.q[1] = *(const I4*)&myP[ml * 32 + 16 + half * 8];
        #pragma unroll
        for (int t = 0; t < 4; ++t) {
            Frag vb;
            const unsigned short* vp = Vt + (t * 16 + ml) * 32;
            vb.q[0] = *(const I4*)(vp + half * 8);
            vb.q[1] = *(const I4*)(vp + 16 + half * 8);
            o[t] = __builtin_amdgcn_wmma_f32_16x16x32_bf16(
                false, pa.bf, false, vb.bf, (short)0, o[t], false, false);
        }
        __syncthreads();
    }

    // ---- normalize, store to [B*L, D] bf16 for out-projection ----
    const int h = bh & (HEADS - 1), b = bh >> 4;
    #pragma unroll
    for (int t = 0; t < 4; ++t) {
        #pragma unroll
        for (int r = 0; r < 8; ++r) {
            const int R = b * SEQ + q0w + r + 8 * half;
            const int col = h * HD + t * 16 + ml;
            attn[(size_t)R * DIM + col] = f2bf(o[t][r] / rsum[r]);
        }
    }
}

// ------------------------------------------------------ out-projection ------
__global__ __launch_bounds__(128) void k_out(const unsigned short* __restrict__ ab,
                                             const unsigned short* __restrict__ wT,
                                             const float* __restrict__ bias,
                                             float* __restrict__ out) {
    __shared__ __align__(16) unsigned short ldsA[2][64 * 32];
    __shared__ __align__(16) unsigned short ldsB[2][64 * 32];
    const int m0 = blockIdx.y * 64, n0 = blockIdx.x * 64;
    const int tid = threadIdx.x;
    const int w = tid >> 5, lane = tid & 31, ml = lane & 15, half = lane >> 4;
    const bool issuer = (__builtin_amdgcn_readfirstlane(tid) >> 5) == 0;

    v8f acc[4];
    #pragma unroll
    for (int t = 0; t < 4; ++t)
        #pragma unroll
        for (int r = 0; r < 8; ++r) acc[t][r] = 0.f;

    if (issuer) {
        tdm_load2d_bf16(lds_off(&ldsA[0][0]), ab + (size_t)m0 * DIM, 32, 64, DIM);
        tdm_load2d_bf16(lds_off(&ldsB[0][0]), wT + (size_t)n0 * DIM, 32, 64, DIM);
    }

    for (int it = 0; it < DIM / 32; ++it) {
        const int cur = it & 1;
        if (issuer) {
            if (it + 1 < DIM / 32) {
                const int kn = (it + 1) * 32;
                tdm_load2d_bf16(lds_off(&ldsA[cur ^ 1][0]), ab + (size_t)m0 * DIM + kn, 32, 64, DIM);
                tdm_load2d_bf16(lds_off(&ldsB[cur ^ 1][0]), wT + (size_t)n0 * DIM + kn, 32, 64, DIM);
                __builtin_amdgcn_s_wait_tensorcnt(2);
            } else {
                __builtin_amdgcn_s_wait_tensorcnt(0);
            }
        }
        __syncthreads();

        Frag a;
        const unsigned short* A = &ldsA[cur][0];
        const unsigned short* B = &ldsB[cur][0];
        const int arow = w * 16 + ml;
        a.q[0] = *(const I4*)&A[arow * 32 + half * 8];
        a.q[1] = *(const I4*)&A[arow * 32 + 16 + half * 8];
        #pragma unroll
        for (int t = 0; t < 4; ++t) {
            Frag b;
            const int brow = t * 16 + ml;
            b.q[0] = *(const I4*)&B[brow * 32 + half * 8];
            b.q[1] = *(const I4*)&B[brow * 32 + 16 + half * 8];
            acc[t] = __builtin_amdgcn_wmma_f32_16x16x32_bf16(
                false, a.bf, false, b.bf, (short)0, acc[t], false, false);
        }
        __syncthreads();
    }

    #pragma unroll
    for (int t = 0; t < 4; ++t) {
        const int Cc = n0 + t * 16 + ml;
        const float bv = bias[Cc];
        #pragma unroll
        for (int r = 0; r < 8; ++r) {
            const int R = m0 + w * 16 + r + 8 * half;
            out[(size_t)R * DIM + Cc] = acc[t][r] + bv;
        }
    }
}

// ---------------------------------------------------------------------------
extern "C" void kernel_launch(void* const* d_in, const int* in_sizes, int n_in,
                              void* d_out, int out_size, void* d_ws, size_t ws_size,
                              hipStream_t stream) {
    const float* x     = (const float*)d_in[0];
    const float* w_qkv = (const float*)d_in[1];
    const float* b_qkv = (const float*)d_in[2];
    const float* w_out = (const float*)d_in[3];
    const float* b_out = (const float*)d_in[4];
    float* out = (float*)d_out;

    char* ws = (char*)d_ws;
    unsigned short* xb    = (unsigned short*)(ws);                       //  8 MB  x bf16
    unsigned short* wqkvT = (unsigned short*)(ws + (size_t)(8  << 20));  //  6 MB  w_qkv^T bf16
    unsigned short* woutT = (unsigned short*)(ws + (size_t)(14 << 20));  //  2 MB  w_out^T bf16
    unsigned short* qb    = (unsigned short*)(ws + (size_t)(16 << 20));  //  8 MB  Q [B,H,L,64]
    unsigned short* kbuf  = (unsigned short*)(ws + (size_t)(24 << 20));  //  8 MB  K [B,H,L,64]
    unsigned short* vT    = (unsigned short*)(ws + (size_t)(32 << 20));  //  8 MB  V^T [B,H,64,L]
    unsigned short* attn  = (unsigned short*)(ws + (size_t)(40 << 20));  //  8 MB  attn out bf16

    k_cvt  <<<(ROWS * DIM) / 256,    256, 0, stream>>>(x, xb, ROWS * DIM);
    k_cvt_t<<<(3 * DIM * DIM) / 256, 256, 0, stream>>>(w_qkv, wqkvT, 3 * DIM, DIM);
    k_cvt_t<<<(DIM * DIM) / 256,     256, 0, stream>>>(w_out, woutT, DIM, DIM);

    k_qkv <<<dim3(3 * DIM / 64, ROWS / 64), 128, 0, stream>>>(xb, wqkvT, b_qkv, qb, kbuf, vT);
    k_attn<<<dim3(SEQ / 64, BATCH * HEADS), 128, 0, stream>>>(qb, kbuf, vT, attn);
    k_out <<<dim3(DIM / 64, ROWS / 64),     128, 0, stream>>>(attn, woutT, b_out, out);
}
